// VNTransformerEncoder_12343736009202
// MI455X (gfx1250) — matrix operand order repaired and checked by reading, not compiled, not run on hardware
//
#include <hip/hip_runtime.h>
#include <hip/hip_bf16.h>

#define EPSF   1e-6f
#define BNEPS  1e-5f

typedef __attribute__((ext_vector_type(16))) _Float16 v16h;
typedef __attribute__((ext_vector_type(2)))  _Float16 h2v;
typedef __attribute__((ext_vector_type(8)))  float    v8f;

// ---------------------------------------------------------------------------
// WMMA helpers (CDNA5 wave32, V_WMMA_F32_16X16X32_F16)
// ---------------------------------------------------------------------------
__device__ __forceinline__ v8f wmma_f16(v16h a, v16h b, v8f c) {
  // 8 args: (neg_a, A, neg_b, B, c_mod, C, reuse_a, reuse_b)
  return __builtin_amdgcn_wmma_f32_16x16x32_f16(false, a, false, b, (short)0, c,
                                                false, false);
}
// A-matrix 16x32 f16: lanes 0-15 hold K {0..7,16..23}, lanes 16-31 {8..15,24..31}
__device__ __forceinline__ int akj(int hi, int j) {
  return ((j & 8) << 1) + hi * 8 + (j & 7);
}
// B-matrix 32x16 f16: lanes 0-15 hold K {0..7,16..23}, lanes 16-31 {8..15,24..31}
__device__ __forceinline__ int bkj(int hi, int j) { return ((j & 8) << 1) + hi * 8 + (j & 7); }

// ---------------------------------------------------------------------------
// vn_linear GEMM, specialized: M % 64 == 0, K % 32 == 0, NC % 64 == 0.
// Block = 4 waves; wave w computes rows [by*64 + w*16, +16) x 64 cols.
// X tile staged in LDS as packed f16 (k,k+1) pairs: converted once, each
// B-fragment dword read yields two elements with no inner-loop cvt.
// ---------------------------------------------------------------------------
__global__ __launch_bounds__(128) void gemm_vnlin(
    const float* __restrict__ W, const float* __restrict__ X,
    const float* __restrict__ R, float* __restrict__ Y,
    int M, int K, int NC, long long sX, long long sY) {
  const int b = blockIdx.z;
  const float* Xb = X + (size_t)b * sX;
  float*       Yb = Y + (size_t)b * sY;
  const float* Rb = R ? R + (size_t)b * sY : (const float*)0;
  const int t = threadIdx.x;
  const int wave = t >> 5;
  const int lane = t & 31;
  const int mr = lane & 15, hi = lane >> 4;
  const int m0 = (blockIdx.y * 4 + wave) * 16;
  const int n0 = blockIdx.x * 64;
  const float* Wrow = W + (size_t)(m0 + mr) * K;

  // xs[r][c] holds halves (k = k0+2r, k0+2r+1) of column n0+c.
  // Row stride 66 dwords; the hi half-wave reads rows +4, 264 dwords further on.
  __shared__ h2v xs[16][66];

  v8f acc[4] = {};
  for (int k0 = 0; k0 < K; k0 += 32) {
    // ---- cooperative staging: 256 tasks (16 k-pairs x 16 col4), 2/thread ----
#pragma unroll
    for (int q = 0; q < 2; ++q) {
      int task = q * 128 + t;  // 0..255
      int r    = task >> 4;    // k-pair row
      int c4   = task & 15;    // group of 4 columns
      const float4* s0 = (const float4*)(Xb + (size_t)(k0 + 2 * r) * NC + n0);
      float4 ve = s0[c4];                  // k even
      float4 vo = s0[(NC >> 2) + c4];      // k odd
      h2v p0; p0.x = (_Float16)ve.x; p0.y = (_Float16)vo.x;
      h2v p1; p1.x = (_Float16)ve.y; p1.y = (_Float16)vo.y;
      h2v p2; p2.x = (_Float16)ve.z; p2.y = (_Float16)vo.z;
      h2v p3; p3.x = (_Float16)ve.w; p3.y = (_Float16)vo.w;
      xs[r][c4 * 4 + 0] = p0;
      xs[r][c4 * 4 + 1] = p1;
      xs[r][c4 * 4 + 2] = p2;
      xs[r][c4 * 4 + 3] = p3;
    }
    if (k0 + 32 < K)  // speculative prefetch of next tile (global_prefetch_b8)
      __builtin_prefetch(Xb + (size_t)(k0 + 32) * NC + n0 + (t & 31) * 4, 0, 0);
    __syncthreads();
    // ---- A fragment: 4x float4 from contiguous W row ----
    v16h a;
    {
      const float4* wr = (const float4*)(Wrow + k0);
      float4 w0 = wr[hi * 2];
      float4 w1 = wr[hi * 2 + 1];
      float4 w2 = wr[4 + hi * 2];
      float4 w3 = wr[4 + hi * 2 + 1];
      a[0]  = (_Float16)w0.x; a[1]  = (_Float16)w0.y;
      a[2]  = (_Float16)w0.z; a[3]  = (_Float16)w0.w;
      a[4]  = (_Float16)w1.x; a[5]  = (_Float16)w1.y;
      a[6]  = (_Float16)w1.z; a[7]  = (_Float16)w1.w;
      a[8]  = (_Float16)w2.x; a[9]  = (_Float16)w2.y;
      a[10] = (_Float16)w2.z; a[11] = (_Float16)w2.w;
      a[12] = (_Float16)w3.x; a[13] = (_Float16)w3.y;
      a[14] = (_Float16)w3.z; a[15] = (_Float16)w3.w;
    }
    // ---- B fragments from packed-f16 LDS: 8 dword reads each, no cvt ----
#pragma unroll
    for (int i = 0; i < 4; ++i) {
      v16h bf;
      int col = i * 16 + mr;
#pragma unroll
      for (int p = 0; p < 8; ++p) {  // element pair (2p,2p+1) <-> k pair
        h2v v = xs[hi * 4 + p + (p & 4)][col];
        bf[2 * p]     = v.x;
        bf[2 * p + 1] = v.y;
      }
      acc[i] = wmma_f16(a, bf, acc[i]);
    }
    __syncthreads();
  }
#pragma unroll
  for (int i = 0; i < 4; ++i) {
#pragma unroll
    for (int r = 0; r < 8; ++r) {
      int m = m0 + r + hi * 8;
      size_t idx = (size_t)m * NC + n0 + i * 16 + mr;
      float v = acc[i][r];
      if (Rb) v += Rb[idx];
      Yb[idx] = v;
    }
  }
}

// pos layer (in-channels == 1): pure outer product Y[b,o,j] = W[o] * X[b,j]
__global__ void outer1(const float* __restrict__ W, const float* __restrict__ X,
                       float* __restrict__ Y, int M, int NC3, int total) {
  int idx = blockIdx.x * blockDim.x + threadIdx.x;
  if (idx >= total) return;
  int j = idx % NC3;
  int o = (idx / NC3) % M;
  int b = idx / (NC3 * M);
  Y[idx] = W[o] * X[(size_t)b * NC3 + j];
}

// Wlin (3 out rows): one thread per (b, col), coalesced over cols
__global__ void gemm_m3(const float* __restrict__ W, const float* __restrict__ X,
                        float* __restrict__ Y, int K, int NC, int total) {
  int idx = blockIdx.x * blockDim.x + threadIdx.x;
  if (idx >= total) return;
  int col = idx % NC;
  int b = idx / NC;
  const float* Xb = X + (size_t)b * K * NC;
  float a0 = 0.f, a1 = 0.f, a2 = 0.f;
  for (int c = 0; c < K; ++c) {
    float x = Xb[(size_t)c * NC + col];
    a0 += W[c] * x;
    a1 += W[K + c] * x;
    a2 += W[2 * K + c] * x;
  }
  float* Yb = Y + (size_t)b * 3 * NC;
  Yb[col] = a0;
  Yb[NC + col] = a1;
  Yb[2 * NC + col] = a2;
}

// ---------------------------------------------------------------------------
// Flash attention: one wave per (b, head, 16-query tile). Head dim 48.
// Branch-free fragment loads (compile-time zeros for A tail; clamped address
// + 0/1 mask for K B-fragments). Online softmax over 32-key blocks.
// ---------------------------------------------------------------------------
__global__ __launch_bounds__(32) void flash_attn(
    const float* __restrict__ Q, const float* __restrict__ K,
    const float* __restrict__ V, float* __restrict__ O, int N) {
  const int b = blockIdx.z, h = blockIdx.y, qt = blockIdx.x;
  const size_t hb = ((size_t)b * 384 + (size_t)h * 48) * N;
  const float* Qh = Q + hb;
  const float* Kh = K + hb;
  const float* Vh = V + hb;
  float*       Oh = O + hb;
  const int lane = threadIdx.x;
  const int nl = lane & 15, hi = lane >> 4;
  const int q0 = qt * 16;

  // Q A-fragments; ks==1, j>=8 -> f>=48 (compile-time zero after unroll)
  v16h aq[2];
#pragma unroll
  for (int ks = 0; ks < 2; ++ks) {
#pragma unroll
    for (int j = 0; j < 16; ++j) {
      float v = 0.f;
      if (ks == 0 || j < 8) {
        int f = ks * 32 + akj(hi, j);
        v = Qh[(size_t)f * N + q0 + nl] * 0.25f;  // 1/sqrt(16)
      }
      aq[ks][j] = (_Float16)v;
    }
  }
  float rm[8], rl[8];
#pragma unroll
  for (int r = 0; r < 8; ++r) { rm[r] = -1e30f; rl[r] = 0.f; }
  v8f oacc[3] = {};
  __shared__ float pbuf[16 * 32];

  for (int kt = 0; kt < N; kt += 32) {
    v8f s0 = {}, s1 = {};
#pragma unroll
    for (int ks = 0; ks < 2; ++ks) {
      // ks==1: f<48 iff j<8 (A is zero for j>=8); branchless mask zeroes all hi==1 lanes
      const float km = (ks == 1 && hi == 1) ? 0.f : 1.f;
      v16h bk0, bk1;
#pragma unroll
      for (int j = 0; j < 16; ++j) {
        int f = ks * 32 + bkj(hi, j);
        int fc = (f < 48) ? f : (f - 32);  // clamped valid row, v_cndmask
        bk0[j] = (_Float16)(Kh[(size_t)fc * N + kt + nl] * km);
        bk1[j] = (_Float16)(Kh[(size_t)fc * N + kt + 16 + nl] * km);
      }
      s0 = wmma_f16(aq[ks], bk0, s0);
      s1 = wmma_f16(aq[ks], bk1, s1);
    }
    // Online softmax: rows 0-7 in lanes 0-15, rows 8-15 in lanes 16-31;
    // xor-shuffles (masks 1..8) reduce within each 16-lane half.
#pragma unroll
    for (int r = 0; r < 8; ++r) {
      float mx = fmaxf(s0[r], s1[r]);
#pragma unroll
      for (int msk = 1; msk < 16; msk <<= 1) mx = fmaxf(mx, __shfl_xor(mx, msk, 32));
      float mnew = fmaxf(rm[r], mx);
      float corr = __expf(rm[r] - mnew);
      rm[r] = mnew;
      float p0 = __expf(s0[r] - mnew);
      float p1 = __expf(s1[r] - mnew);
      s0[r] = p0; s1[r] = p1;
      float rs = p0 + p1;
#pragma unroll
      for (int msk = 1; msk < 16; msk <<= 1) rs += __shfl_xor(rs, msk, 32);
      rl[r] = rl[r] * corr + rs;
      oacc[0][r] *= corr; oacc[1][r] *= corr; oacc[2][r] *= corr;
    }
    // Stage P (C-fragment layout) -> LDS -> A-fragment layout
#pragma unroll
    for (int r = 0; r < 8; ++r) {
      int m = hi * 8 + r;
      pbuf[m * 32 + nl]      = s0[r];
      pbuf[m * 32 + 16 + nl] = s1[r];
    }
    __syncthreads();
    v16h pa;
#pragma unroll
    for (int j = 0; j < 16; ++j) pa[j] = (_Float16)pbuf[nl * 32 + akj(hi, j)];
    __syncthreads();
#pragma unroll
    for (int dt = 0; dt < 3; ++dt) {  // d = dt*16 + nl < 48 always: no checks
      v16h bv;
#pragma unroll
      for (int j = 0; j < 16; ++j) {
        int key = kt + bkj(hi, j);
        int dd = dt * 16 + nl;
        bv[j] = (_Float16)Vh[(size_t)dd * N + key];
      }
      oacc[dt] = wmma_f16(pa, bv, oacc[dt]);
    }
  }
#pragma unroll
  for (int dt = 0; dt < 3; ++dt) {
#pragma unroll
    for (int r = 0; r < 8; ++r) {
      int m = hi * 8 + r;
      int dd = dt * 16 + nl;
      Oh[(size_t)dd * N + q0 + m] = oacc[dt][r] / rl[r];
    }
  }
}

// ---------------------------------------------------------------------------
// VN batch-norm over vector norms: reduce (per channel) then apply.
// ---------------------------------------------------------------------------
__global__ void bn_reduce(const float* __restrict__ Xv, float* __restrict__ mu,
                          float* __restrict__ var, int C, int N, int B) {
  const int c = blockIdx.x;
  const int tid = threadIdx.x;
  float s = 0.f, sq = 0.f;
  for (int idx = tid; idx < B * N; idx += blockDim.x) {
    int b = idx / N, n = idx - b * N;
    size_t base = (((size_t)b * C + c) * 3) * N + n;
    float x0 = Xv[base], x1 = Xv[base + N], x2 = Xv[base + 2 * N];
    float nn = sqrtf(x0 * x0 + x1 * x1 + x2 * x2) + EPSF;
    s += nn; sq += nn * nn;
  }
  __shared__ float ss[256], ssq[256];
  ss[tid] = s; ssq[tid] = sq;
  __syncthreads();
  for (int d = 128; d > 0; d >>= 1) {
    if (tid < d) { ss[tid] += ss[tid + d]; ssq[tid] += ssq[tid + d]; }
    __syncthreads();
  }
  if (tid == 0) {
    float m = ss[0] / (float)(B * N);
    mu[c] = m;
    var[c] = ssq[0] / (float)(B * N) - m * m;
  }
}

__global__ void bn_apply(float* __restrict__ Xv, const float* __restrict__ mu,
                         const float* __restrict__ var, const float* __restrict__ g,
                         const float* __restrict__ be, int C, int N, int total) {
  int idx = blockIdx.x * blockDim.x + threadIdx.x;
  if (idx >= total) return;
  int n = idx % N;
  int c = (idx / N) % C;
  int b = idx / (N * C);
  size_t base = (((size_t)b * C + c) * 3) * N + n;
  float x0 = Xv[base], x1 = Xv[base + N], x2 = Xv[base + 2 * N];
  float nn = sqrtf(x0 * x0 + x1 * x1 + x2 * x2) + EPSF;
  float nb = (nn - mu[c]) * rsqrtf(var[c] + BNEPS) * g[c] + be[c];
  float sc = nb / nn;
  Xv[base] = x0 * sc; Xv[base + N] = x1 * sc; Xv[base + 2 * N] = x2 * sc;
}

// vn_lrelu epilogue: out = q - (1-slope) * min(dot,0)/(|d|^2+eps) * d
__global__ void vn_lrelu_post(const float* __restrict__ Qv, const float* __restrict__ Dv,
                              float* __restrict__ Ov, int C, int N, int total, float slope) {
  int idx = blockIdx.x * blockDim.x + threadIdx.x;
  if (idx >= total) return;
  int n = idx % N;
  int c = (idx / N) % C;
  int b = idx / (N * C);
  size_t base = (((size_t)b * C + c) * 3) * N + n;
  float q0 = Qv[base], q1 = Qv[base + N], q2 = Qv[base + 2 * N];
  float d0 = Dv[base], d1 = Dv[base + N], d2 = Dv[base + 2 * N];
  float dot = q0 * d0 + q1 * d1 + q2 * d2;
  float dsq = d0 * d0 + d1 * d1 + d2 * d2;
  float coef = (dot >= 0.f) ? 0.f : ((1.f - slope) * dot / (dsq + EPSF));
  Ov[base]         = q0 - coef * d0;
  Ov[base + N]     = q1 - coef * d1;
  Ov[base + 2 * N] = q2 - coef * d2;
}

// ---------------------------------------------------------------------------
// misc small kernels
// ---------------------------------------------------------------------------
__global__ void x_to_h(const float* __restrict__ X, float* __restrict__ H,
                       int N, int total) {
  int idx = blockIdx.x * blockDim.x + threadIdx.x;
  if (idx >= total) return;
  int n = idx % N;
  int d = (idx / N) % 3;
  int b = idx / (3 * N);
  H[idx] = X[((size_t)b * N + n) * 3 + d];
}

__global__ void mean_n(const float* __restrict__ H, float* __restrict__ xm,
                       int C, int N) {
  int bid = blockIdx.x;  // (b*C + c)*3 + i
  const float* p = H + (size_t)bid * N;
  float s = 0.f;
  for (int n = threadIdx.x; n < N; n += blockDim.x) s += p[n];
  __shared__ float sm[128];
  sm[threadIdx.x] = s;
  __syncthreads();
  for (int d = 64; d > 0; d >>= 1) {
    if ((int)threadIdx.x < d) sm[threadIdx.x] += sm[threadIdx.x + d];
    __syncthreads();
  }
  if (threadIdx.x == 0) xm[bid] = sm[0] / (float)N;
}

__global__ void build_xc(const float* __restrict__ H, const float* __restrict__ xm,
                         float* __restrict__ XC, int C, int N, int total) {
  int idx = blockIdx.x * blockDim.x + threadIdx.x;
  if (idx >= total) return;
  int n  = idx % N;
  int i  = (idx / N) % 3;
  int ch = (idx / (3 * N)) % (2 * C);
  int b  = idx / (3 * N * 2 * C);
  float v;
  if (ch < C) v = H[(((size_t)b * C + ch) * 3 + i) * N + n];
  else        v = xm[((size_t)b * C + (ch - C)) * 3 + i];
  XC[idx] = v;
}

// xs = xc(2C x 3) * z0_swapped(3 x 3) per point; Vp = max over points
__global__ void xs_max(const float* __restrict__ XC, const float* __restrict__ Z0,
                       float* __restrict__ Vf, int C2, int N) {
  const int i = blockIdx.x % C2, b = blockIdx.x / C2;
  float mx[3] = {-1e30f, -1e30f, -1e30f};
  for (int m = threadIdx.x; m < N; m += blockDim.x) {
    float xc0 = XC[(((size_t)b * C2 + i) * 3 + 0) * N + m];
    float xc1 = XC[(((size_t)b * C2 + i) * 3 + 1) * N + m];
    float xc2 = XC[(((size_t)b * C2 + i) * 3 + 2) * N + m];
#pragma unroll
    for (int k = 0; k < 3; ++k) {
      float l0 = Z0[(((size_t)b * 3 + k) * 3 + 0) * N + m];
      float l1 = Z0[(((size_t)b * 3 + k) * 3 + 1) * N + m];
      float l2 = Z0[(((size_t)b * 3 + k) * 3 + 2) * N + m];
      mx[k] = fmaxf(mx[k], xc0 * l0 + xc1 * l1 + xc2 * l2);
    }
  }
  __shared__ float sm[3][128];
#pragma unroll
  for (int k = 0; k < 3; ++k) sm[k][threadIdx.x] = mx[k];
  __syncthreads();
  for (int d = 64; d > 0; d >>= 1) {
    if ((int)threadIdx.x < d) {
#pragma unroll
      for (int k = 0; k < 3; ++k)
        sm[k][threadIdx.x] = fmaxf(sm[k][threadIdx.x], sm[k][threadIdx.x + d]);
    }
    __syncthreads();
  }
  if (threadIdx.x == 0) {
#pragma unroll
    for (int k = 0; k < 3; ++k) Vf[(size_t)b * 784 + i * 3 + k] = sm[k][0];
  }
}

// ---------------------------------------------------------------------------
// Geometric (knn) branch
// ---------------------------------------------------------------------------
__global__ void zero_stats(float* s) {
  if (threadIdx.x < 2) s[threadIdx.x] = 0.f;
}

__global__ __launch_bounds__(32) void knn_kernel(const float* __restrict__ X,
                                                 float* __restrict__ knn,
                                                 float* __restrict__ stats, int N) {
  const int row = blockIdx.x;
  const int b = row / N, n = row % N;
  const int lane = threadIdx.x;
  const float px = X[((size_t)b * N + n) * 3 + 0];
  const float py = X[((size_t)b * N + n) * 3 + 1];
  const float pz = X[((size_t)b * N + n) * 3 + 2];
  float best[20];
#pragma unroll
  for (int i = 0; i < 20; ++i) best[i] = 1e30f;
  for (int j = lane; j < N; j += 32) {
    float dx = px - X[((size_t)b * N + j) * 3 + 0];
    float dy = py - X[((size_t)b * N + j) * 3 + 1];
    float dz = pz - X[((size_t)b * N + j) * 3 + 2];
    float d = sqrtf(fmaxf(dx * dx + dy * dy + dz * dz, 1e-12f));
    if (d < best[19]) {
      float v = d;
#pragma unroll
      for (int p = 0; p < 20; ++p) {  // bubble insert, static indices
        float cv = best[p];
        if (v < cv) { best[p] = v; v = cv; }
      }
    }
  }
  __shared__ float lds[32 * 20];
  __shared__ int head[32];
#pragma unroll
  for (int i = 0; i < 20; ++i) lds[lane * 20 + i] = best[i];
  head[lane] = 0;
  __syncthreads();
  if (lane == 0) {  // merge 32 sorted lists, pop 20 smallest
    float s = 0.f, sq = 0.f;
    for (int sel = 0; sel < 20; ++sel) {
      float mn = 1e30f; int mi = 0;
      for (int t = 0; t < 32; ++t) {
        if (head[t] < 20) {
          float v = lds[t * 20 + head[t]];
          if (v < mn) { mn = v; mi = t; }
        }
      }
      head[mi]++;
      knn[(size_t)row * 20 + sel] = mn;
      s += mn; sq += mn * mn;
    }
    atomicAdd(&stats[0], s);
    atomicAdd(&stats[1], sq);
  }
}

__global__ void geo_kernel(const float* __restrict__ knn, const float* __restrict__ stats,
                           const float* __restrict__ w, const float* __restrict__ bb,
                           const float* __restrict__ g, const float* __restrict__ be,
                           float* __restrict__ Vf, int N) {
  (void)bb;  // bias cancels inside batch-norm
  const int c = blockIdx.x & 15, b = blockIdx.x >> 4;
  const float cnt = 8.f * (float)N * 20.f;
  const float m = stats[0] / cnt;
  const float v = stats[1] / cnt - m * m;
  const float wc = w[c];
  const float sc = wc * rsqrtf(wc * wc * v + BNEPS);
  const float gg = g[c], bt = be[c];
  float s = 0.f;
  for (int n = threadIdx.x; n < N; n += blockDim.x) {
    float mx = -1e30f;
    const float* kp = knn + ((size_t)b * N + n) * 20;
#pragma unroll
    for (int k = 0; k < 20; ++k) {
      float t = (kp[k] - m) * sc * gg + bt;
      t = (t >= 0.f) ? t : 0.2f * t;
      mx = fmaxf(mx, t);
    }
    s += mx;
  }
  __shared__ float sm[128];
  sm[threadIdx.x] = s;
  __syncthreads();
  for (int d = 64; d > 0; d >>= 1) {
    if ((int)threadIdx.x < d) sm[threadIdx.x] += sm[threadIdx.x + d];
    __syncthreads();
  }
  if (threadIdx.x == 0) Vf[(size_t)b * 784 + 768 + c] = sm[0] / (float)N;
}

// ---------------------------------------------------------------------------
// Heads (tiny: B=8 rows)
// ---------------------------------------------------------------------------
__global__ void inv1(const float* __restrict__ Vf, const float* __restrict__ W1,
                     const float* __restrict__ b1, float* __restrict__ Y) {
  int b = blockIdx.x, o = threadIdx.x;
  const float* vb = Vf + (size_t)b * 784;
  const float* wr = W1 + (size_t)o * 784;
  float s = b1[o];
  for (int k = 0; k < 784; ++k) s += vb[k] * wr[k];
  Y[(size_t)b * 256 + o] = s;
}

__global__ void inv_bn(float* __restrict__ Y, const float* __restrict__ g,
                       const float* __restrict__ be) {
  int c = threadIdx.x;
  float v[8]; float m = 0.f;
#pragma unroll
  for (int b = 0; b < 8; ++b) { v[b] = Y[b * 256 + c]; m += v[b]; }
  m *= 0.125f;
  float vv = 0.f;
#pragma unroll
  for (int b = 0; b < 8; ++b) { float d = v[b] - m; vv += d * d; }
  vv *= 0.125f;
  float sc = rsqrtf(vv + BNEPS);
#pragma unroll
  for (int b = 0; b < 8; ++b) {
    float t = (v[b] - m) * sc * g[c] + be[c];
    Y[b * 256 + c] = (t >= 0.f) ? t : 0.2f * t;
  }
}

__global__ void inv2(const float* __restrict__ Y, const float* __restrict__ W2,
                     const float* __restrict__ b2, float* __restrict__ Z) {
  int b = blockIdx.x, o = threadIdx.x;
  const float* yb = Y + (size_t)b * 256;
  const float* wr = W2 + (size_t)o * 256;
  float s = b2[o];
  for (int k = 0; k < 256; ++k) s += yb[k] * wr[k];
  Z[(size_t)b * 384 + o] = s;
}

__global__ void eq_head(const float* __restrict__ xm, const float* __restrict__ W,
                        float* __restrict__ Z) {
  int b = blockIdx.x, t = threadIdx.x;
  int o = t / 3, d = t % 3;
  float s = 0.f;
  for (int c = 0; c < 128; ++c)
    s += xm[((size_t)b * 128 + c) * 3 + d] * W[(size_t)o * 128 + c];
  Z[(size_t)b * 384 + t] = s;
}

__global__ __launch_bounds__(32) void cry_head(
    const float* __restrict__ zinv, const float* __restrict__ W1,
    const float* __restrict__ b1, const float* __restrict__ W2,
    const float* __restrict__ b2, float* __restrict__ out) {
  int b = blockIdx.x, j = threadIdx.x;
  const float* zb = zinv + (size_t)b * 384;
  const float* wr = W1 + (size_t)j * 384;
  float t = b1[j];
  for (int k = 0; k < 384; ++k) t += zb[k] * wr[k];
  t = (t >= 0.f) ? t : 0.01f * t;
  float s = t * W2[j];
#pragma unroll
  for (int msk = 1; msk < 32; msk <<= 1) s += __shfl_xor(s, msk, 32);
  if (j == 0) out[b] = 1.f / (1.f + __expf(-(s + b2[0])));
}

// ---------------------------------------------------------------------------
// Host orchestration
// ---------------------------------------------------------------------------
extern "C" void kernel_launch(void* const* d_in, const int* in_sizes, int n_in,
                              void* d_out, int out_size, void* d_ws, size_t ws_size,
                              hipStream_t stream) {
  (void)in_sizes; (void)n_in; (void)out_size; (void)ws_size;
  auto F = [&](int i) { return (const float*)d_in[i]; };
  int id = 0;
  const float* x        = F(id++);
  const float* geo_w    = F(id++); const float* geo_b   = F(id++);
  const float* geo_g    = F(id++); const float* geo_be  = F(id++);
  const float* pos_Wf   = F(id++); const float* pos_Wd  = F(id++);
  const float* pos_g    = F(id++); const float* pos_b   = F(id++);
  const float* c1_Wf    = F(id++); const float* c1_Wd   = F(id++);
  const float* c1_g     = F(id++); const float* c1_b    = F(id++);
  struct Layer {
    const float *Wq, *Wk, *Wv, *Wo, *g1, *b1, *fWf, *fWd, *fg, *fb, *Wf2, *g2, *b2;
  } L[4];
  for (int l = 0; l < 4; ++l) {
    L[l].Wq = F(id++); L[l].Wk = F(id++); L[l].Wv = F(id++); L[l].Wo = F(id++);
    L[l].g1 = F(id++); L[l].b1 = F(id++);
    L[l].fWf = F(id++); L[l].fWd = F(id++); L[l].fg = F(id++); L[l].fb = F(id++);
    L[l].Wf2 = F(id++); L[l].g2 = F(id++); L[l].b2 = F(id++);
  }
  const float* v1_Wf = F(id++); const float* v1_Wd = F(id++);
  const float* v1_g  = F(id++); const float* v1_b  = F(id++);
  const float* v2_Wf = F(id++); const float* v2_Wd = F(id++);
  const float* v2_g  = F(id++); const float* v2_b  = F(id++);
  const float* Wlin  = F(id++);
  const float* iW1 = F(id++); const float* ib1  = F(id++);
  const float* ig  = F(id++); const float* ibe  = F(id++);
  const float* iW2 = F(id++); const float* ib2  = F(id++);
  const float* eqW = F(id++);
  const float* cW1 = F(id++); const float* cb1 = F(id++);
  const float* cW2 = F(id++); const float* cb2 = F(id++);

  const int B = 8, N = 1024, NC = 3072;
  float* ws = (float*)d_ws;
  const size_t S = (size_t)B * 256 * 3 * N;  // one full-size vec buffer
  float* bufA = ws;
  float* bufH = ws + S;
  float* bufQ = ws + 2 * S;
  float* bufK = ws + 3 * S;
  float* bufV = ws + 4 * S;
  float* bufT = ws + 5 * S;
  float* small = ws + 6 * S;
  float* knnb  = small;                       // B*N*20
  float* stats = knnb + (size_t)B * N * 20;   // 2
  float* bnmu  = stats + 2;                   // 256
  float* bnvar = bnmu + 256;                  // 256
  float* xmean = bnvar + 256;                 // B*128*3
  float* z0b   = xmean + (size_t)B * 128 * 3; // B*3*3*N
  float* Vf    = z0b + (size_t)B * 9 * N;     // B*784
  float* ybuf  = Vf + (size_t)B * 784;        // B*256

  auto gemm = [&](const float* W, const float* Xv, const float* R, float* Y,
                  int M, int K) {  // requires M%64==0, K%32==0
    dim3 g(NC / 64, M / 64, B);
    gemm_vnlin<<<g, 128, 0, stream>>>(W, Xv, R, Y, M, K, NC,
                                      (long long)K * NC, (long long)M * NC);
  };
  auto bn = [&](float* Xv, const float* g_, const float* b_, int C) {
    bn_reduce<<<C, 256, 0, stream>>>(Xv, bnmu, bnvar, C, N, B);
    int tot = B * C * N;
    bn_apply<<<(tot + 255) / 256, 256, 0, stream>>>(Xv, bnmu, bnvar, g_, b_, C, N, tot);
  };
  auto post = [&](const float* Qv, const float* Dv, float* Ov, int C, float slope) {
    int tot = B * C * N;
    vn_lrelu_post<<<(tot + 255) / 256, 256, 0, stream>>>(Qv, Dv, Ov, C, N, tot, slope);
  };

  // ---- geometric branch ----
  zero_stats<<<1, 32, 0, stream>>>(stats);
  knn_kernel<<<B * N, 32, 0, stream>>>(x, knnb, stats, N);
  geo_kernel<<<B * 16, 128, 0, stream>>>(knnb, stats, geo_w, geo_b, geo_g, geo_be, Vf, N);

  // ---- equivariant trunk ----
  x_to_h<<<(B * 3 * N + 255) / 256, 256, 0, stream>>>(x, bufT, N, B * 3 * N);
  // pos: 1 -> 64 channels (outer product, no GEMM needed)
  {
    int t1 = B * 64 * NC;
    outer1<<<(t1 + 255) / 256, 256, 0, stream>>>(pos_Wf, bufT, bufQ, 64, NC, t1);
    bn(bufQ, pos_g, pos_b, 64);
    outer1<<<(t1 + 255) / 256, 256, 0, stream>>>(pos_Wd, bufT, bufK, 64, NC, t1);
    post(bufQ, bufK, bufH, 64, 0.f);
  }
  // c1: 64 -> 128
  gemm(c1_Wf, bufH, nullptr, bufQ, 128, 64);
  bn(bufQ, c1_g, c1_b, 128);
  gemm(c1_Wd, bufH, nullptr, bufK, 128, 64);
  post(bufQ, bufK, bufT, 128, 0.f);

  float* cur = bufT;
  float* alt = bufH;
  for (int l = 0; l < 4; ++l) {
    // attention
    gemm(L[l].Wq, cur, nullptr, bufQ, 128, 128);
    gemm(L[l].Wk, cur, nullptr, bufK, 128, 128);
    gemm(L[l].Wv, cur, nullptr, bufV, 128, 128);
    flash_attn<<<dim3(N / 16, 8, B), 32, 0, stream>>>(bufQ, bufK, bufV, bufA, N);
    gemm(L[l].Wo, bufA, cur, bufQ, 128, 128);  // + residual
    bn(bufQ, L[l].g1, L[l].b1, 128);
    // ffn
    gemm(L[l].fWf, bufQ, nullptr, bufK, 256, 128);
    bn(bufK, L[l].fg, L[l].fb, 256);
    gemm(L[l].fWd, bufQ, nullptr, bufV, 256, 128);
    post(bufK, bufV, bufK, 256, 0.f);
    gemm(L[l].Wf2, bufK, bufQ, alt, 128, 256);  // + residual
    bn(alt, L[l].g2, L[l].b2, 128);
    float* t = cur; cur = alt; alt = t;
  }

  // ---- VNStdFeature ----
  mean_n<<<B * 128 * 3, 128, 0, stream>>>(cur, xmean, 128, N);
  int totxc = B * 256 * 3 * N;
  build_xc<<<(totxc + 255) / 256, 256, 0, stream>>>(cur, xmean, bufA, 128, N, totxc);
  gemm(v1_Wf, bufA, nullptr, bufQ, 256, 256);
  bn(bufQ, v1_g, v1_b, 256);
  gemm(v1_Wd, bufA, nullptr, bufK, 256, 256);
  post(bufQ, bufK, bufQ, 256, 0.2f);
  gemm(v2_Wf, bufQ, nullptr, bufV, 128, 256);
  bn(bufV, v2_g, v2_b, 128);
  gemm(v2_Wd, bufQ, nullptr, bufH, 128, 256);
  post(bufV, bufH, bufV, 128, 0.2f);
  {
    int t3 = B * NC;
    gemm_m3<<<(t3 + 255) / 256, 256, 0, stream>>>(Wlin, bufV, z0b, 128, NC, t3);
  }
  xs_max<<<B * 256, 128, 0, stream>>>(bufA, z0b, Vf, 256, N);

  // ---- heads ----
  float* out = (float*)d_out;
  inv1<<<B, 256, 0, stream>>>(Vf, iW1, ib1, ybuf);
  inv_bn<<<1, 256, 0, stream>>>(ybuf, ig, ibe);
  inv2<<<B, 384, 0, stream>>>(ybuf, iW2, ib2, out);              // z_inv [8,384]
  eq_head<<<B, 384, 0, stream>>>(xmean, eqW, out + 3072);        // z_eq  [8,128,3]
  cry_head<<<B, 32, 0, stream>>>(out, cW1, cb1, cW2, cb2, out + 6144);  // cr [8,1]
}
